// MultiHeadSelfAttention_68590627717718
// MI455X (gfx1250) — matrix operand order repaired and checked by reading, not compiled, not run on hardware
//
#include <hip/hip_runtime.h>
#include <hip/hip_bf16.h>
#include <stdint.h>

// ---------- types ----------
typedef __attribute__((ext_vector_type(16))) __bf16 v16bf;
typedef __attribute__((ext_vector_type(8)))  __bf16 v8bf;
typedef __attribute__((ext_vector_type(8)))  float  v8f;

union V16 { v16bf v; v8bf h8[2]; };

#define WMMA_BF16(A, B, C) \
  __builtin_amdgcn_wmma_f32_16x16x32_bf16(false, (A), false, (B), (short)0, (C), false, false)

// bf16 hi/lo split helpers (truncation split: x == hi + lo to ~16 mantissa bits)
static __device__ __forceinline__ __bf16 bf_hi(float f) {
  unsigned u = __builtin_bit_cast(unsigned, f);
  return __builtin_bit_cast(__bf16, (unsigned short)(u >> 16));
}
static __device__ __forceinline__ float f_hi(float f) {
  unsigned u = __builtin_bit_cast(unsigned, f) & 0xFFFF0000u;
  return __builtin_bit_cast(float, u);
}
static __device__ __forceinline__ __bf16 bf_lo(float f) {
  return (__bf16)(f - f_hi(f));
}

// ---------- problem constants ----------
#define GM 65536          // B_*N rows
#define GK 1024           // C
#define GN 3072           // 3*C
#define QKV_SECT 67108864 // B_*N*C : size of each of q/k/v sections in flat qkv

// ---------- kernel 1: fp32 -> (bf16 hi, bf16 lo) ----------
__global__ void split_kernel(const float* __restrict__ src,
                             __bf16* __restrict__ hi, __bf16* __restrict__ lo, int n) {
  int idx = blockIdx.x * blockDim.x + threadIdx.x;
  int stride = gridDim.x * blockDim.x;
  for (int i = idx; i < n; i += stride) {
    float f = src[i];
    hi[i] = bf_hi(f);
    lo[i] = bf_lo(f);
  }
}

// ---------- kernel 2: scrambled bias expansion: (225,32) gathered -> (32,64,64) flat ----------
__global__ void bias_kernel(const float* __restrict__ table, const int* __restrict__ relidx,
                            float* __restrict__ bias) {
  int i = blockIdx.x * blockDim.x + threadIdx.x;  // 0..131071
  bias[i] = table[relidx[i >> 5] * 32 + (i & 31)];
}

// ---------- kernel 3: QKV GEMM, C[M,N] = Xsplit * Wsplit^T (fp32 accumulate) ----------
// one wave computes a 64x64 C tile; split bf16: ah*bh + ah*bl + al*bh
__global__ __launch_bounds__(256) void qkv_gemm(const __bf16* __restrict__ Ahi,
                                                const __bf16* __restrict__ Alo,
                                                const __bf16* __restrict__ Bhi,
                                                const __bf16* __restrict__ Blo,
                                                float* __restrict__ C) {
  const int wave = threadIdx.x >> 5;
  const int lane = threadIdx.x & 31;
  const int wid  = blockIdx.x * 8 + wave;   // 0..49151
  const int mTile = wid / 48;               // waves in a block share the A panel
  const int nTile = wid % 48;
  const int mBase = mTile * 64;
  const int nBase = nTile * 64;
  const int lm = lane & 15, lh = lane >> 4;

  v8f acc[4][4] = {};

  for (int kc = 0; kc < GK; kc += 32) {
    V16 ah[4], al[4], bh[4], bl[4];
    // A fragments: 16x32 bf16; lane half picks K octets {0-7,16-23} / {8-15,24-31}
    #pragma unroll
    for (int t = 0; t < 4; ++t) {
      const __bf16* pA = Ahi + (size_t)(mBase + 16 * t + lm) * GK + kc + lh * 8;
      const __bf16* pAl = Alo + (pA - Ahi);
      ah[t].h8[0] = *(const v8bf*)pA;
      ah[t].h8[1] = *(const v8bf*)(pA + 16);
      al[t].h8[0] = *(const v8bf*)pAl;
      al[t].h8[1] = *(const v8bf*)(pAl + 16);
      __builtin_prefetch(pA + 32, 0, 3);   // next K-chunk
    }
    // B fragments: 32x16 bf16; lane = column (W row), lane half picks K octets {0-7,16-23} / {8-15,24-31}
    #pragma unroll
    for (int j = 0; j < 4; ++j) {
      const __bf16* pB = Bhi + (size_t)(nBase + 16 * j + lm) * GK + kc + lh * 8;
      const __bf16* pBl = Blo + (pB - Bhi);
      bh[j].h8[0] = *(const v8bf*)pB;
      bh[j].h8[1] = *(const v8bf*)(pB + 16);
      bl[j].h8[0] = *(const v8bf*)pBl;
      bl[j].h8[1] = *(const v8bf*)(pBl + 16);
      __builtin_prefetch(pB + 32, 0, 3);
    }
    #pragma unroll
    for (int t = 0; t < 4; ++t)
      #pragma unroll
      for (int j = 0; j < 4; ++j) {
        acc[t][j] = WMMA_BF16(ah[t].v, bh[j].v, acc[t][j]);
        acc[t][j] = WMMA_BF16(ah[t].v, bl[j].v, acc[t][j]);
        acc[t][j] = WMMA_BF16(al[t].v, bh[j].v, acc[t][j]);
      }
  }

  // store: acc element (r,lane) -> C[mBase+16t+r+8*lh][nBase+16j+lm]
  #pragma unroll
  for (int t = 0; t < 4; ++t)
    #pragma unroll
    for (int j = 0; j < 4; ++j)
      #pragma unroll
      for (int r = 0; r < 8; ++r) {
        size_t row = (size_t)(mBase + 16 * t + r + 8 * lh);
        C[row * GN + nBase + 16 * j + lm] = acc[t][j][r];
      }
}

// ---------- kernel 4: attention per (window b2, head h); one wave each ----------
__global__ __launch_bounds__(256) void attn_kernel(const float* __restrict__ qkv,
                                                   const float* __restrict__ bias,
                                                   float* __restrict__ out) {
  __shared__ __align__(16) __bf16 pHi[8][16 * 64];
  __shared__ __align__(16) __bf16 pLo[8][16 * 64];

  const int wave = threadIdx.x >> 5;
  const int lane = threadIdx.x & 31;
  const int pair = blockIdx.x * 8 + wave;   // 0..32767
  const int b2 = pair >> 5;
  const int hh = pair & 31;
  const int lm = lane & 15, lh = lane >> 4;
  const float SCALE = 0.17677669529663687f; // 32^-0.5

  const float* Q  = qkv + (size_t)b2 * 65536 + (size_t)hh * 2048;
  const float* Kp = Q + QKV_SECT;
  const float* Vp = Kp + QKV_SECT;
  const float* biasH = bias + hh * 4096;

  // K fragments (B operand of S=Q*K^T), register-resident across row blocks.
  // lane = score column m = 16j+lm; lane half = d half (16 contiguous floats).
  V16 kfh[4], kfl[4];
  #pragma unroll
  for (int j = 0; j < 4; ++j) {
    const float* p = Kp + (16 * j + lm) * 32 + lh * 16;
    #pragma unroll
    for (int e = 0; e < 16; ++e) {
      float f = p[e];
      kfh[j].v[e] = bf_hi(f);
      kfl[j].v[e] = bf_lo(f);
    }
  }
  // V fragments (B operand of O=P*V): lane = out column d = 16jd+lm; K(=m) gathered stride 32.
  V16 vfh[2][2], vfl[2][2];
  #pragma unroll
  for (int jd = 0; jd < 2; ++jd)
    #pragma unroll
    for (int c = 0; c < 2; ++c)
      #pragma unroll
      for (int e = 0; e < 16; ++e) {
        int m = 32 * c + lh * 8 + (e & 7) + (e >> 3) * 16;
        float f = Vp[m * 32 + 16 * jd + lm];
        vfh[jd][c].v[e] = bf_hi(f);
        vfl[jd][c].v[e] = bf_lo(f);
      }

  for (int i = 0; i < 4; ++i) {
    // Q A-fragment for row block i (pre-scaled)
    V16 qh, ql;
    const float* pQ = Q + (16 * i + lm) * 32;
    #pragma unroll
    for (int e = 0; e < 8; ++e) {
      float f0 = pQ[lh * 16 + e] * SCALE;
      float f1 = pQ[lh * 16 + 8 + e] * SCALE;
      qh.v[e] = bf_hi(f0); ql.v[e] = bf_lo(f0);
      qh.v[8 + e] = bf_hi(f1); ql.v[8 + e] = bf_lo(f1);
    }
    // scores: init accumulators with bias, accumulate scaled Q*K^T on top
    v8f s[4];
    #pragma unroll
    for (int j = 0; j < 4; ++j)
      #pragma unroll
      for (int r = 0; r < 8; ++r)
        s[j][r] = biasH[(16 * i + r + 8 * lh) * 64 + 16 * j + lm];
    #pragma unroll
    for (int j = 0; j < 4; ++j) {
      s[j] = WMMA_BF16(qh.v, kfh[j].v, s[j]);
      s[j] = WMMA_BF16(qh.v, kfl[j].v, s[j]);
      s[j] = WMMA_BF16(ql.v, kfh[j].v, s[j]);
    }
    // softmax per row (row M=r+8*lh lives in one 16-lane half; 4 elements per lane)
    #pragma unroll
    for (int r = 0; r < 8; ++r) {
      float mx = fmaxf(fmaxf(s[0][r], s[1][r]), fmaxf(s[2][r], s[3][r]));
      mx = fmaxf(mx, __shfl_xor(mx, 1, 32));
      mx = fmaxf(mx, __shfl_xor(mx, 2, 32));
      mx = fmaxf(mx, __shfl_xor(mx, 4, 32));
      mx = fmaxf(mx, __shfl_xor(mx, 8, 32));
      float e0 = __expf(s[0][r] - mx);
      float e1 = __expf(s[1][r] - mx);
      float e2 = __expf(s[2][r] - mx);
      float e3 = __expf(s[3][r] - mx);
      float sum = (e0 + e1) + (e2 + e3);
      sum += __shfl_xor(sum, 1, 32);
      sum += __shfl_xor(sum, 2, 32);
      sum += __shfl_xor(sum, 4, 32);
      sum += __shfl_xor(sum, 8, 32);
      float rs = 1.0f / sum;
      int row = r + 8 * lh;
      float p0 = e0 * rs, p1 = e1 * rs, p2 = e2 * rs, p3 = e3 * rs;
      pHi[wave][row * 64 +  0 + lm] = bf_hi(p0); pLo[wave][row * 64 +  0 + lm] = bf_lo(p0);
      pHi[wave][row * 64 + 16 + lm] = bf_hi(p1); pLo[wave][row * 64 + 16 + lm] = bf_lo(p1);
      pHi[wave][row * 64 + 32 + lm] = bf_hi(p2); pLo[wave][row * 64 + 32 + lm] = bf_lo(p2);
      pHi[wave][row * 64 + 48 + lm] = bf_hi(p3); pLo[wave][row * 64 + 48 + lm] = bf_lo(p3);
    }
    asm volatile("s_wait_dscnt 0" ::: "memory"); // cross-lane LDS transpose visibility

    // reload P in A-operand layout (lane row = lm, K octets per lane half)
    V16 ph[2], pl[2];
    #pragma unroll
    for (int c = 0; c < 2; ++c) {
      const __bf16* bh = &pHi[wave][lm * 64 + c * 32 + lh * 8];
      const __bf16* bl = &pLo[wave][lm * 64 + c * 32 + lh * 8];
      ph[c].h8[0] = *(const v8bf*)bh;
      ph[c].h8[1] = *(const v8bf*)(bh + 16);
      pl[c].h8[0] = *(const v8bf*)bl;
      pl[c].h8[1] = *(const v8bf*)(bl + 16);
    }
    asm volatile("s_wait_dscnt 0" ::: "memory"); // loads done before next-iter overwrite

    // O = P * V
    v8f o[2] = {};
    #pragma unroll
    for (int c = 0; c < 2; ++c)
      #pragma unroll
      for (int jd = 0; jd < 2; ++jd) {
        o[jd] = WMMA_BF16(ph[c].v, vfh[jd][c].v, o[jd]);
        o[jd] = WMMA_BF16(ph[c].v, vfl[jd][c].v, o[jd]);
        o[jd] = WMMA_BF16(pl[c].v, vfh[jd][c].v, o[jd]);
      }
    // store out[b2, n, hh*32 + d]
    #pragma unroll
    for (int jd = 0; jd < 2; ++jd)
      #pragma unroll
      for (int r = 0; r < 8; ++r) {
        int n = 16 * i + r + 8 * lh;
        out[((size_t)b2 * 64 + n) * 1024 + hh * 32 + 16 * jd + lm] = o[jd][r];
      }
  }
}

// ---------- launch ----------
extern "C" void kernel_launch(void* const* d_in, const int* in_sizes, int n_in,
                              void* d_out, int out_size, void* d_ws, size_t ws_size,
                              hipStream_t stream) {
  const float* x      = (const float*)d_in[0];  // (1024,64,1024)
  const float* W_qkv  = (const float*)d_in[1];  // (3072,1024)
  const float* table  = (const float*)d_in[2];  // (225,32)
  const int*   relidx = (const int*)d_in[3];    // (64,64)
  float* out = (float*)d_out;

  char* ws = (char*)d_ws;
  size_t off = 0;
  float*  qkv     = (float*)(ws + off);  off += (size_t)GM * GN * 4;       // 805,306,368
  float*  biasExp = (float*)(ws + off);  off += 131072 * 4;                // 524,288
  __bf16* Xhi     = (__bf16*)(ws + off); off += (size_t)GM * GK * 2;       // 134,217,728
  __bf16* Xlo     = (__bf16*)(ws + off); off += (size_t)GM * GK * 2;
  __bf16* Whi     = (__bf16*)(ws + off); off += (size_t)GN * GK * 2;       // 6,291,456
  __bf16* Wlo     = (__bf16*)(ws + off); off += (size_t)GN * GK * 2;

  split_kernel<<<2048, 256, 0, stream>>>(x, Xhi, Xlo, GM * GK);
  split_kernel<<<512, 256, 0, stream>>>(W_qkv, Whi, Wlo, GN * GK);
  bias_kernel<<<512, 256, 0, stream>>>(table, relidx, biasExp);
  qkv_gemm<<<6144, 256, 0, stream>>>(Xhi, Xlo, Whi, Wlo, qkv);
  attn_kernel<<<4096, 256, 0, stream>>>(qkv, biasExp, out);
}
